// TiledLinear_12730283065838
// MI455X (gfx1250) — compile-verified, matrix operand order checked by reading, not run on hardware
//
#include <hip/hip_runtime.h>
#include <hip/hip_bf16.h>

// out[t, o] = sum_k X[t,k] * W[o,k] + bias[o]
// M=4096 (4*1024 tokens), K=2048, N=2048, fp32 in/out, bf16 WMMA compute.
//
// Fast path: fp32 -> bf16 pre-pass into d_ws, then a 128x256-tile GEMM whose
// LDS is filled with gfx1250 async copies (global_load_async_to_lds_b128,
// drained by s_wait_asynccnt) and consumed by v_wmma_f32_16x16x32_bf16.

typedef __attribute__((ext_vector_type(8)))  float  v8f;
typedef __attribute__((ext_vector_type(8)))  __bf16 v8bf;
typedef __attribute__((ext_vector_type(16))) __bf16 v16bf;

#define M_DIM 4096
#define N_DIM 2048
#define K_DIM 2048

// ---------------- shared helpers ----------------

__device__ __forceinline__ v8bf cvt_pack8(float4 a, float4 b) {
  v8bf r;
  r[0] = (__bf16)a.x; r[1] = (__bf16)a.y; r[2] = (__bf16)a.z; r[3] = (__bf16)a.w;
  r[4] = (__bf16)b.x; r[5] = (__bf16)b.y; r[6] = (__bf16)b.z; r[7] = (__bf16)b.w;
  return r;
}

// Per-lane fragment gather matching the ISA 16-bit A/B layout:
// lane<16  : K = {0..7, 16..23} of its row
// lane>=16 : K = {8..15, 24..31} of its row
__device__ __forceinline__ v16bf load_frag(const __bf16* rowbase, int hi8) {
  v8bf c0 = *(const v8bf*)(rowbase + hi8);
  v8bf c1 = *(const v8bf*)(rowbase + 16 + hi8);
  return __builtin_shufflevector(c0, c1, 0, 1, 2, 3, 4, 5, 6, 7,
                                         8, 9, 10, 11, 12, 13, 14, 15);
}

// Raw LDS byte offset of a __shared__ object (VDST operand of async loads).
__device__ __forceinline__ uint32_t lds_off_of(const void* p) {
  return (uint32_t)(uintptr_t)(const __attribute__((address_space(3))) void*)p;
}

// gfx1250 async copy: 16 bytes global -> LDS per lane, tracked by ASYNCcnt.
__device__ __forceinline__ void async_copy_b128(uint32_t lds_byte_off, const void* gaddr) {
  asm volatile("global_load_async_to_lds_b128 %0, %1, off"
               :: "v"(lds_byte_off), "v"(gaddr)
               : "memory");
}

__device__ __forceinline__ void wait_async0() {
  asm volatile("s_wait_asynccnt 0x0" ::: "memory");
}

// ---------------- pass 1: fp32 -> bf16 ----------------

__global__ __launch_bounds__(256)
void cvt_f32_bf16_kernel(const float* __restrict__ src, __bf16* __restrict__ dst) {
  const size_t i = ((size_t)blockIdx.x * blockDim.x + threadIdx.x) * 8;
  float4 a = *(const float4*)(src + i);
  float4 b = *(const float4*)(src + i + 4);
  *(v8bf*)(dst + i) = cvt_pack8(a, b);
}

// ---------------- fast path GEMM: bf16 ws + async LDS fill ----------------

constexpr int BM2   = 128;
constexpr int BN2   = 256;
constexpr int BK2   = 32;
constexpr int LDSS2 = 40;             // padded row stride in bf16 elems (80 B)
constexpr int NK2   = K_DIM / BK2;    // 64 K-steps

__global__ __launch_bounds__(256)
void gemm_bf16_async_wmma(const __bf16* __restrict__ Xb,
                          const __bf16* __restrict__ Wb,
                          const float*  __restrict__ bias,
                          float* __restrict__ out) {
  __shared__ __bf16 sA[2][BM2 * LDSS2];   // 2 x 10240 B
  __shared__ __bf16 sB[2][BN2 * LDSS2];   // 2 x 20480 B  (total 60 KB)

  const int tid   = threadIdx.x;
  const int lane  = tid & 31;
  const int wave  = tid >> 5;
  const int laneR = lane & 15;
  const int laneH = lane >> 4;
  const int hi8   = laneH * 8;

  const int m0 = blockIdx.y * BM2;
  const int n0 = blockIdx.x * BN2;

  // 8 waves: 2 in M x 4 in N; each wave computes 64(M) x 64(N) = 4x4 fragments.
  const int wm = (wave >> 2) * 64;
  const int wn = (wave & 3) * 64;

  // Async fill assignment (16B chunks): chunk c -> row c>>2, col (c&3)*8 elems.
  // A tile: 512 chunks (2/thread). B tile: 1024 chunks (4/thread).
  const int ca0 = tid, ca1 = tid + 256;
  const uint32_t a_lds0 = lds_off_of(&sA[0][(ca0 >> 2) * LDSS2 + (ca0 & 3) * 8]);
  const uint32_t a_lds1 = lds_off_of(&sA[0][(ca1 >> 2) * LDSS2 + (ca1 & 3) * 8]);
  uint32_t b_lds[4];
  const __bf16* a_g[2];
  const __bf16* b_g[4];
#pragma unroll
  for (int j = 0; j < 4; ++j) {
    const int c = tid + 256 * j;
    b_lds[j] = lds_off_of(&sB[0][(c >> 2) * LDSS2 + (c & 3) * 8]);
    b_g[j]   = Wb + (size_t)(n0 + (c >> 2)) * K_DIM + (c & 3) * 8;
  }
  a_g[0] = Xb + (size_t)(m0 + (ca0 >> 2)) * K_DIM + (ca0 & 3) * 8;
  a_g[1] = Xb + (size_t)(m0 + (ca1 >> 2)) * K_DIM + (ca1 & 3) * 8;

  constexpr uint32_t A_BUF_STRIDE = BM2 * LDSS2 * 2;  // bytes per A buffer
  constexpr uint32_t B_BUF_STRIDE = BN2 * LDSS2 * 2;  // bytes per B buffer

  // Prologue: async-load tile 0 into buffer 0.
  async_copy_b128(a_lds0, a_g[0]);
  async_copy_b128(a_lds1, a_g[1]);
#pragma unroll
  for (int j = 0; j < 4; ++j) async_copy_b128(b_lds[j], b_g[j]);

  v8f acc[4][4] = {};

  for (int kt = 0; kt < NK2; ++kt) {
    const int buf = kt & 1;

    wait_async0();       // this wave's async fills for `buf` are in LDS
    __syncthreads();     // everyone's fills visible; prior-buf reads all done

    // Kick off next tile into the other buffer; overlaps the WMMAs below.
    if (kt + 1 < NK2) {
      const int koff = (kt + 1) * BK2;
      const uint32_t ab = (kt + 1) & 1 ? A_BUF_STRIDE : 0;
      const uint32_t bb = (kt + 1) & 1 ? B_BUF_STRIDE : 0;
      async_copy_b128(a_lds0 + ab, a_g[0] + koff);
      async_copy_b128(a_lds1 + ab, a_g[1] + koff);
#pragma unroll
      for (int j = 0; j < 4; ++j) async_copy_b128(b_lds[j] + bb, b_g[j] + koff);
    }

    v16bf afrag[4];
#pragma unroll
    for (int mi = 0; mi < 4; ++mi)
      afrag[mi] = load_frag(&sA[buf][(wm + mi * 16 + laneR) * LDSS2], hi8);

#pragma unroll
    for (int ni = 0; ni < 4; ++ni) {
      v16bf bfrag = load_frag(&sB[buf][(wn + ni * 16 + laneR) * LDSS2], hi8);
#pragma unroll
      for (int mi = 0; mi < 4; ++mi)
        acc[mi][ni] = __builtin_amdgcn_wmma_f32_16x16x32_bf16(
            false, afrag[mi], false, bfrag, (short)0, acc[mi][ni], false, false);
    }
  }

  // Epilogue: C/D layout is VGPR v -> M = v (+8 for lanes 16..31), N = lane%16.
#pragma unroll
  for (int ni = 0; ni < 4; ++ni) {
    const int col = n0 + wn + ni * 16 + laneR;
    const float bv = bias[col];
#pragma unroll
    for (int mi = 0; mi < 4; ++mi) {
      const int rowb = m0 + wm + mi * 16 + laneH * 8;
#pragma unroll
      for (int v = 0; v < 8; ++v)
        out[(size_t)(rowb + v) * N_DIM + col] = acc[mi][ni][v] + bv;
    }
  }
}

// ---------------- fallback: round-1 kernel (fp32 in, cvt in-loop) ----------------

constexpr int BM1   = 128;
constexpr int BN1   = 128;
constexpr int BK1   = 32;
constexpr int LDSS1 = 40;
constexpr int NK1   = K_DIM / BK1;

__global__ __launch_bounds__(256)
void gemm_f32_direct_wmma(const float* __restrict__ X,
                          const float* __restrict__ W,
                          const float* __restrict__ bias,
                          float* __restrict__ out) {
  __shared__ __bf16 sA[2][BM1 * LDSS1];
  __shared__ __bf16 sB[2][BN1 * LDSS1];

  const int tid   = threadIdx.x;
  const int lane  = tid & 31;
  const int wave  = tid >> 5;
  const int laneR = lane & 15;
  const int laneH = lane >> 4;
  const int hi8   = laneH * 8;

  const int m0 = blockIdx.y * BM1;
  const int n0 = blockIdx.x * BN1;
  const int wm = (wave >> 1) * 32;
  const int wn = (wave & 1) * 64;

  const int gRow  = tid >> 1;
  const int gHalf = (tid & 1) * 16;
  const float* Ag = X + (size_t)(m0 + gRow) * K_DIM + gHalf;
  const float* Bg = W + (size_t)(n0 + gRow) * K_DIM + gHalf;

  float4 ra[4], rb[4];
#pragma unroll
  for (int j = 0; j < 4; ++j) {
    ra[j] = *(const float4*)(Ag + j * 4);
    rb[j] = *(const float4*)(Bg + j * 4);
  }

  v8f acc[2][4] = {};

  for (int kt = 0; kt < NK1; ++kt) {
    const int buf = kt & 1;
    {
      __bf16* da = &sA[buf][gRow * LDSS1 + gHalf];
      __bf16* db = &sB[buf][gRow * LDSS1 + gHalf];
      *(v8bf*)(da)     = cvt_pack8(ra[0], ra[1]);
      *(v8bf*)(da + 8) = cvt_pack8(ra[2], ra[3]);
      *(v8bf*)(db)     = cvt_pack8(rb[0], rb[1]);
      *(v8bf*)(db + 8) = cvt_pack8(rb[2], rb[3]);
    }
    __syncthreads();

    if (kt + 1 < NK1) {
      const float* Ag2 = Ag + (kt + 1) * BK1;
      const float* Bg2 = Bg + (kt + 1) * BK1;
#pragma unroll
      for (int j = 0; j < 4; ++j) {
        ra[j] = *(const float4*)(Ag2 + j * 4);
        rb[j] = *(const float4*)(Bg2 + j * 4);
      }
    }

    v16bf afrag[2], bfrag[4];
#pragma unroll
    for (int mi = 0; mi < 2; ++mi)
      afrag[mi] = load_frag(&sA[buf][(wm + mi * 16 + laneR) * LDSS1], hi8);
#pragma unroll
    for (int ni = 0; ni < 4; ++ni)
      bfrag[ni] = load_frag(&sB[buf][(wn + ni * 16 + laneR) * LDSS1], hi8);

#pragma unroll
    for (int mi = 0; mi < 2; ++mi)
#pragma unroll
      for (int ni = 0; ni < 4; ++ni)
        acc[mi][ni] = __builtin_amdgcn_wmma_f32_16x16x32_bf16(
            false, afrag[mi], false, bfrag[ni], (short)0, acc[mi][ni], false, false);
  }

#pragma unroll
  for (int ni = 0; ni < 4; ++ni) {
    const int col = n0 + wn + ni * 16 + laneR;
    const float bv = bias[col];
#pragma unroll
    for (int mi = 0; mi < 2; ++mi) {
      const int rowb = m0 + wm + mi * 16 + laneH * 8;
#pragma unroll
      for (int v = 0; v < 8; ++v)
        out[(size_t)(rowb + v) * N_DIM + col] = acc[mi][ni][v] + bv;
    }
  }
}

// ---------------- launch ----------------

extern "C" void kernel_launch(void* const* d_in, const int* in_sizes, int n_in,
                              void* d_out, int out_size, void* d_ws, size_t ws_size,
                              hipStream_t stream) {
  (void)in_sizes; (void)n_in; (void)out_size;
  const float* X  = (const float*)d_in[0];   // (4,1024,2048) fp32
  const float* W  = (const float*)d_in[1];   // (2048,2048) fp32
  const float* Bi = (const float*)d_in[2];   // (2048,) fp32
  float* out      = (float*)d_out;           // (4,1024,2048) fp32

  const size_t xElems = (size_t)M_DIM * K_DIM;           // 8,388,608
  const size_t wElems = (size_t)N_DIM * K_DIM;           // 4,194,304
  const size_t need   = (xElems + wElems) * sizeof(__bf16);

  if (ws_size >= need) {
    __bf16* Xb = (__bf16*)d_ws;
    __bf16* Wb = Xb + xElems;
    cvt_f32_bf16_kernel<<<dim3((unsigned)(xElems / 8 / 256)), dim3(256), 0, stream>>>(X, Xb);
    cvt_f32_bf16_kernel<<<dim3((unsigned)(wElems / 8 / 256)), dim3(256), 0, stream>>>(W, Wb);
    dim3 grid(N_DIM / BN2, M_DIM / BM2);   // 8 x 32
    gemm_bf16_async_wmma<<<grid, dim3(256), 0, stream>>>(Xb, Wb, Bi, out);
  } else {
    dim3 grid(N_DIM / BN1, M_DIM / BM1);   // 16 x 32
    gemm_f32_direct_wmma<<<grid, dim3(256), 0, stream>>>(X, W, Bi, out);
  }
}
